// Encoder_75634374083175
// MI455X (gfx1250) — compile-verified, matrix operand order checked by reading, not run on hardware
//
#include <hip/hip_runtime.h>
#include <hip/hip_bf16.h>
#include <math.h>

typedef __bf16 bf16_t;
typedef __attribute__((ext_vector_type(16))) __bf16 v16bf;
typedef __attribute__((ext_vector_type(8)))  float   v8f;
typedef __attribute__((ext_vector_type(4)))  unsigned int u32x4;
typedef __attribute__((ext_vector_type(8)))  unsigned int u32x8;

#define NB     4
#define SEQ    2048
#define MTOK   (NB * SEQ)        // 8192
#define DMODEL 768
#define NHEAD  12
#define HDIM   64
#define DQKV   (3 * DMODEL)      // 2304
#define DFFN   (4 * DMODEL)      // 3072
#define LNEPS  1e-5f

__device__ __forceinline__ bf16_t f2bf(float x) { return (bf16_t)x; }

__device__ __forceinline__ v8f wmma_bf16(v16bf a, v16bf b, v8f c) {
  return __builtin_amdgcn_wmma_f32_16x16x32_bf16(
      false, a, false, b, (short)0, c, false, false);
}

// ---- Tensor Data Mover: 2-D tile (tile_cols x tile_rows) of bf16 into LDS ---
// D# group0: count=1 | lds_addr | global_addr | type=2 ("image")
// D# group1: data_size=2B, tensor_dim0=row_pitch, tile_dim0=tile_cols,
//            tile_dim1=tile_rows, tensor_dim0_stride=row_pitch
__device__ __forceinline__ void tdm_load_2d(unsigned int lds_off,
                                            const bf16_t* gptr,
                                            unsigned int tile_cols,
                                            unsigned int tile_rows,
                                            unsigned int row_pitch_elems) {
  unsigned long long ga = (unsigned long long)(size_t)gptr;
  u32x4 g0;
  g0[0] = 1u;                                               // count=1, user mode
  g0[1] = lds_off;                                          // LDS byte address
  g0[2] = (unsigned int)ga;                                 // global_addr[31:0]
  g0[3] = (unsigned int)((ga >> 32) & 0x1FFFFFFu) | (2u << 30);  // addr hi | type=2
  const unsigned int td0 = row_pitch_elems;                 // tensor_dim0
  const unsigned int td1 = 0x00100000u;                     // tensor_dim1 (large)
  u32x8 g1;
  g1[0] = (1u << 16);                                       // data_size=1 (2 bytes)
  g1[1] = (td0 & 0xFFFFu) << 16;                            // tensor_dim0 lo
  g1[2] = ((td0 >> 16) & 0xFFFFu) | ((td1 & 0xFFFFu) << 16);
  g1[3] = ((td1 >> 16) & 0xFFFFu) | (tile_cols << 16);      // tile_dim0
  g1[4] = tile_rows & 0xFFFFu;                              // tile_dim1, tile_dim2=0
  g1[5] = row_pitch_elems;                                  // tensor_dim0_stride lo
  g1[6] = 0u;
  g1[7] = 0u;
  asm volatile("tensor_load_to_lds %0, %1" :: "s"(g0), "s"(g1) : "memory");
}

__device__ __forceinline__ void wait_tensorcnt0() {
  asm volatile("s_wait_tensorcnt 0x0" ::: "memory");
}

// A-style fragment of a 16x32 tile from a row-major source (lane = row).
__device__ __forceinline__ v16bf load_frag_row(const bf16_t* __restrict__ p0,
                                               int ld, int k0) {
  const int lane = threadIdx.x & 31;
  const bf16_t* q = p0 + (size_t)(lane & 15) * ld + k0 + ((lane >> 4) << 3);
  v16bf f;
#pragma unroll
  for (int j = 0; j < 8; ++j) { f[j] = q[j]; f[j + 8] = q[j + 16]; }
  return f;
}

// B fragment of a 32x16 (KxN) tile from a row-major [K x ld] source (lane=col).
__device__ __forceinline__ v16bf load_frag_col(const bf16_t* __restrict__ p0,
                                               int ld, int k0) {
  const int lane = threadIdx.x & 31;
  const bf16_t* q = p0 + (size_t)(k0 + ((lane >> 4) << 3)) * ld + (lane & 15);
  v16bf f;
#pragma unroll
  for (int j = 0; j < 8; ++j) { f[j] = q[(size_t)j * ld]; f[j + 8] = q[(size_t)(j + 16) * ld]; }
  return f;
}

// ---------------------------------------------------------------- convert ---
__global__ __launch_bounds__(256) void cvt_f32_bf16(const float* __restrict__ in,
                                                    bf16_t* __restrict__ out,
                                                    int n) {
  for (int i = blockIdx.x * blockDim.x + threadIdx.x; i < n;
       i += gridDim.x * blockDim.x)
    out[i] = f2bf(in[i]);
}

// ------------------------------------------------------------------- GEMM ---
// C[M,N] = A[M,K] * B[N,K]^T. Block 64x128 (8 waves, wave tile 32x32).
// A/B k-slices DMA'd to LDS by the TDM, double buffered.
enum { FUSE_BF16_OUT = 0, FUSE_RES_F32 = 1, FUSE_GELU_BF16 = 2 };

template <int FUSE>
__global__ __launch_bounds__(256) void wmma_gemm_kernel(
    const bf16_t* __restrict__ A, const bf16_t* __restrict__ B,
    float* __restrict__ outF, bf16_t* __restrict__ outB,
    const float* __restrict__ res, int M, int N, int K) {
  __shared__ __align__(16) bf16_t sA[2][64 * 32];
  __shared__ __align__(16) bf16_t sB[2][128 * 32];
  const int wave = threadIdx.x >> 5;
  const int lane = threadIdx.x & 31;
  const int wm = wave >> 2, wn = wave & 3;
  const int row0 = blockIdx.x * 64;
  const int col0 = blockIdx.y * 128;

  v8f acc[2][2];
#pragma unroll
  for (int a = 0; a < 2; ++a)
#pragma unroll
    for (int b = 0; b < 2; ++b) acc[a][b] = (v8f){};

  const bf16_t* Ap = A + (size_t)row0 * K;
  const bf16_t* Bp = B + (size_t)col0 * K;
  const int nk = K / 32;

  if (wave == 0) {
    tdm_load_2d((unsigned int)(size_t)&sA[0][0], Ap, 32, 64, K);
    tdm_load_2d((unsigned int)(size_t)&sB[0][0], Bp, 32, 128, K);
  }
  for (int ks = 0; ks < nk; ++ks) {
    const int cur = ks & 1;
    if (wave == 0) wait_tensorcnt0();
    __syncthreads();                         // buffers[cur] ready for everyone
    if (wave == 0 && ks + 1 < nk) {
      tdm_load_2d((unsigned int)(size_t)&sA[cur ^ 1][0], Ap + (ks + 1) * 32, 32, 64, K);
      tdm_load_2d((unsigned int)(size_t)&sB[cur ^ 1][0], Bp + (ks + 1) * 32, 32, 128, K);
    }
    const bf16_t* tA = &sA[cur][wm * 32 * 32];
    const bf16_t* tB = &sB[cur][wn * 32 * 32];
    v16bf a0 = load_frag_row(tA, 32, 0);
    v16bf a1 = load_frag_row(tA + 16 * 32, 32, 0);
    v16bf b0 = load_frag_row(tB, 32, 0);
    v16bf b1 = load_frag_row(tB + 16 * 32, 32, 0);
    acc[0][0] = wmma_bf16(a0, b0, acc[0][0]);
    acc[0][1] = wmma_bf16(a0, b1, acc[0][1]);
    acc[1][0] = wmma_bf16(a1, b0, acc[1][0]);
    acc[1][1] = wmma_bf16(a1, b1, acc[1][1]);
    __syncthreads();                         // done reading buffers[cur]
  }

  const int g = lane >> 4, cl = lane & 15;
#pragma unroll
  for (int mi = 0; mi < 2; ++mi)
#pragma unroll
    for (int ni = 0; ni < 2; ++ni)
#pragma unroll
      for (int i = 0; i < 8; ++i) {
        int r = row0 + wm * 32 + mi * 16 + i + 8 * g;
        int c = col0 + wn * 32 + ni * 16 + cl;
        size_t idx = (size_t)r * N + c;
        float x = acc[mi][ni][i];
        if (FUSE == FUSE_BF16_OUT) {
          outB[idx] = f2bf(x);
        } else if (FUSE == FUSE_RES_F32) {
          outF[idx] = x + res[idx];
        } else {  // exact GELU
          outB[idx] = f2bf(0.5f * x * (1.0f + erff(x * 0.70710678118654752f)));
        }
      }
}

// -------------------------------------------------------- flash attention ---
// One block = 8 waves = 8 consecutive 16-row q-tiles of the SAME (n,h);
// K/V 32-key slices are TDM-staged into LDS and shared by all waves.
__global__ __launch_bounds__(256) void flash_attn_kernel(
    const bf16_t* __restrict__ qkv,   // [MTOK, DQKV]: 0 Q | 768 K | 1536 V
    const int* __restrict__ mask,     // [NB, SEQ], 0 = padded
    bf16_t* __restrict__ ctx) {       // [MTOK, DMODEL]
  __shared__ __align__(16) bf16_t sK[2][32 * HDIM];
  __shared__ __align__(16) bf16_t sV[2][32 * HDIM];
  __shared__ __align__(16) bf16_t plds[8][16][32];
  const int wave = threadIdx.x >> 5;
  const int lane = threadIdx.x & 31;
  const int bid = blockIdx.x;
  const int qg = bid & 15;                    // 16 groups of 8 q-tiles
  const int h  = (bid >> 4) % NHEAD;
  const int n  = bid / (16 * NHEAD);
  const int qt = qg * 8 + wave;

  const bf16_t* Qp = qkv + (size_t)(n * SEQ + qt * 16) * DQKV + h * HDIM;
  const bf16_t* Kb = qkv + (size_t)(n * SEQ) * DQKV + DMODEL + h * HDIM;
  const bf16_t* Vb = qkv + (size_t)(n * SEQ) * DQKV + 2 * DMODEL + h * HDIM;
  const int* mrow = mask + n * SEQ;

  const v16bf aq0 = load_frag_row(Qp, DQKV, 0);
  const v16bf aq1 = load_frag_row(Qp, DQKV, 32);

  v8f o[4];
#pragma unroll
  for (int t = 0; t < 4; ++t) o[t] = (v8f){};
  float m[8], l[8];
#pragma unroll
  for (int i = 0; i < 8; ++i) { m[i] = -3.0e38f; l[i] = 0.0f; }

  const int g = lane >> 4, cl = lane & 15;

  if (wave == 0) {
    tdm_load_2d((unsigned int)(size_t)&sK[0][0], Kb, HDIM, 32, DQKV);
    tdm_load_2d((unsigned int)(size_t)&sV[0][0], Vb, HDIM, 32, DQKV);
  }
  for (int kb = 0; kb < SEQ; kb += 32) {
    const int cur = (kb >> 5) & 1;
    if (wave == 0) wait_tensorcnt0();
    __syncthreads();
    if (wave == 0 && kb + 32 < SEQ) {
      tdm_load_2d((unsigned int)(size_t)&sK[cur ^ 1][0], Kb + (size_t)(kb + 32) * DQKV, HDIM, 32, DQKV);
      tdm_load_2d((unsigned int)(size_t)&sV[cur ^ 1][0], Vb + (size_t)(kb + 32) * DQKV, HDIM, 32, DQKV);
    }

    // scores: two 16x16 tiles over 32 keys, K-dim = 64 (2 chained wmma each)
    v8f s0 = (v8f){}, s1 = (v8f){};
    s0 = wmma_bf16(aq0, load_frag_row(&sK[cur][0], HDIM, 0), s0);
    s0 = wmma_bf16(aq1, load_frag_row(&sK[cur][0], HDIM, 32), s0);
    s1 = wmma_bf16(aq0, load_frag_row(&sK[cur][16 * HDIM], HDIM, 0), s1);
    s1 = wmma_bf16(aq1, load_frag_row(&sK[cur][16 * HDIM], HDIM, 32), s1);

    const float msk0 = (mrow[kb + cl] == 0) ? -1.0e30f : 0.0f;
    const float msk1 = (mrow[kb + 16 + cl] == 0) ? -1.0e30f : 0.0f;

#pragma unroll
    for (int i = 0; i < 8; ++i) {
      float x0 = s0[i] * 0.125f + msk0;     // 1/sqrt(64)
      float x1 = s1[i] * 0.125f + msk1;
      float v = fmaxf(x0, x1);
      v = fmaxf(v, __shfl_xor(v, 8, 32));
      v = fmaxf(v, __shfl_xor(v, 4, 32));
      v = fmaxf(v, __shfl_xor(v, 2, 32));
      v = fmaxf(v, __shfl_xor(v, 1, 32));
      float mnew = fmaxf(m[i], v);
      float corr = __expf(m[i] - mnew);
      float p0 = __expf(x0 - mnew);
      float p1 = __expf(x1 - mnew);
      float ps = p0 + p1;
      ps += __shfl_xor(ps, 8, 32);
      ps += __shfl_xor(ps, 4, 32);
      ps += __shfl_xor(ps, 2, 32);
      ps += __shfl_xor(ps, 1, 32);
      l[i] = l[i] * corr + ps;
      m[i] = mnew;
#pragma unroll
      for (int t = 0; t < 4; ++t) o[t][i] = o[t][i] * corr;
      plds[wave][i + 8 * g][cl]      = f2bf(p0);
      plds[wave][i + 8 * g][16 + cl] = f2bf(p1);
    }
    asm volatile("s_wait_dscnt 0x0" ::: "memory");
    v16bf pf = load_frag_row(&plds[wave][0][0], 32, 0);
#pragma unroll
    for (int t = 0; t < 4; ++t) {
      v16bf vf = load_frag_col(&sV[cur][t * 16], HDIM, 0);
      o[t] = wmma_bf16(pf, vf, o[t]);
    }
    __syncthreads();                         // done reading sK/sV[cur]
  }

  const size_t rowb = (size_t)(n * SEQ + qt * 16);
#pragma unroll
  for (int i = 0; i < 8; ++i) {
    float inv = 1.0f / l[i];
    size_t r = rowb + i + 8 * g;
#pragma unroll
    for (int t = 0; t < 4; ++t)
      ctx[r * DMODEL + h * HDIM + t * 16 + cl] = f2bf(o[t][i] * inv);
  }
}

// -------------------------------------------------------------- layernorm ---
__global__ __launch_bounds__(256) void ln_kernel(
    const float* __restrict__ in, const float* __restrict__ gamma,
    const float* __restrict__ beta, float* __restrict__ outF,
    bf16_t* __restrict__ outB) {
  __shared__ float red[256];
  const int t = threadIdx.x;
  const float* x = in + (size_t)blockIdx.x * DMODEL;
  float v0 = x[t], v1 = x[t + 256], v2 = x[t + 512];

  red[t] = v0 + v1 + v2;
  __syncthreads();
  for (int off = 128; off > 0; off >>= 1) {
    if (t < off) red[t] += red[t + off];
    __syncthreads();
  }
  float mu = red[0] * (1.0f / DMODEL);
  __syncthreads();

  float d0 = v0 - mu, d1 = v1 - mu, d2 = v2 - mu;
  red[t] = d0 * d0 + d1 * d1 + d2 * d2;
  __syncthreads();
  for (int off = 128; off > 0; off >>= 1) {
    if (t < off) red[t] += red[t + off];
    __syncthreads();
  }
  float rstd = rsqrtf(red[0] * (1.0f / DMODEL) + LNEPS);

  float* yF = outF + (size_t)blockIdx.x * DMODEL;
  bf16_t* yB = outB ? outB + (size_t)blockIdx.x * DMODEL : nullptr;
  float y0 = d0 * rstd * gamma[t] + beta[t];
  float y1 = d1 * rstd * gamma[t + 256] + beta[t + 256];
  float y2 = d2 * rstd * gamma[t + 512] + beta[t + 512];
  yF[t] = y0; yF[t + 256] = y1; yF[t + 512] = y2;
  if (yB) { yB[t] = f2bf(y0); yB[t + 256] = f2bf(y1); yB[t + 512] = f2bf(y2); }
}

// ------------------------------------------------------------------ driver ---
extern "C" void kernel_launch(void* const* d_in, const int* in_sizes, int n_in,
                              void* d_out, int out_size, void* d_ws,
                              size_t ws_size, hipStream_t stream) {
  const float* x    = (const float*)d_in[0];
  const int*   mask = (const int*)d_in[1];
  const float* Wq   = (const float*)d_in[2];
  const float* Wk   = (const float*)d_in[3];
  const float* Wv   = (const float*)d_in[4];
  const float* Wo   = (const float*)d_in[5];
  const float* W1   = (const float*)d_in[6];
  const float* W2   = (const float*)d_in[7];
  const float* lng  = (const float*)d_in[8];
  const float* lnb  = (const float*)d_in[9];

  char* p = (char*)d_ws;
  auto carve = [&](size_t bytes) -> void* {
    void* r = (void*)p;
    p += (bytes + 255) & ~(size_t)255;
    return r;
  };
  bf16_t* Xb    = (bf16_t*)carve((size_t)MTOK * DMODEL * 2);
  bf16_t* Wqkvb = (bf16_t*)carve((size_t)DQKV * DMODEL * 2);
  bf16_t* Wob   = (bf16_t*)carve((size_t)DMODEL * DMODEL * 2);
  bf16_t* W1b   = (bf16_t*)carve((size_t)DFFN * DMODEL * 2);
  bf16_t* W2b   = (bf16_t*)carve((size_t)DMODEL * DFFN * 2);
  bf16_t* QKVb  = (bf16_t*)carve((size_t)MTOK * DQKV * 2);
  bf16_t* CTXb  = (bf16_t*)carve((size_t)MTOK * DMODEL * 2);
  float*  H1    = (float*)carve((size_t)MTOK * DMODEL * 4);
  float*  N1f   = (float*)carve((size_t)MTOK * DMODEL * 4);
  bf16_t* N1b   = (bf16_t*)carve((size_t)MTOK * DMODEL * 2);
  bf16_t* Gb    = (bf16_t*)carve((size_t)MTOK * DFFN * 2);
  float*  H2    = (float*)carve((size_t)MTOK * DMODEL * 4);

  const dim3 blk(256);
  const int DD = DMODEL * DMODEL;

  cvt_f32_bf16<<<2048, blk, 0, stream>>>(x, Xb, MTOK * DMODEL);
  cvt_f32_bf16<<<512, blk, 0, stream>>>(Wq, Wqkvb, DD);
  cvt_f32_bf16<<<512, blk, 0, stream>>>(Wk, Wqkvb + DD, DD);
  cvt_f32_bf16<<<512, blk, 0, stream>>>(Wv, Wqkvb + 2 * DD, DD);
  cvt_f32_bf16<<<512, blk, 0, stream>>>(Wo, Wob, DD);
  cvt_f32_bf16<<<1024, blk, 0, stream>>>(W1, W1b, DFFN * DMODEL);
  cvt_f32_bf16<<<1024, blk, 0, stream>>>(W2, W2b, DMODEL * DFFN);

  // QKV = X * Wqkv^T (bf16 out)
  wmma_gemm_kernel<FUSE_BF16_OUT><<<dim3(MTOK / 64, DQKV / 128), blk, 0, stream>>>(
      Xb, Wqkvb, nullptr, QKVb, nullptr, MTOK, DQKV, DMODEL);

  // flash attention -> ctx (bf16)
  flash_attn_kernel<<<dim3(NB * NHEAD * 16), blk, 0, stream>>>(QKVb, mask, CTXb);

  // H1 = ctx * Wo^T + x
  wmma_gemm_kernel<FUSE_RES_F32><<<dim3(MTOK / 64, DMODEL / 128), blk, 0, stream>>>(
      CTXb, Wob, H1, nullptr, x, MTOK, DMODEL, DMODEL);

  ln_kernel<<<MTOK, blk, 0, stream>>>(H1, lng, lnb, N1f, N1b);

  // G = gelu(n1 * W1^T) (bf16 out)
  wmma_gemm_kernel<FUSE_GELU_BF16><<<dim3(MTOK / 64, DFFN / 128), blk, 0, stream>>>(
      N1b, W1b, nullptr, Gb, nullptr, MTOK, DFFN, DMODEL);

  // H2 = G * W2^T + n1
  wmma_gemm_kernel<FUSE_RES_F32><<<dim3(MTOK / 64, DMODEL / 128), blk, 0, stream>>>(
      Gb, W2b, H2, nullptr, N1f, MTOK, DMODEL, DFFN);

  ln_kernel<<<MTOK, blk, 0, stream>>>(H2, lng, lnb, (float*)d_out, nullptr);
}
